// LSTMModel_68204080661051
// MI455X (gfx1250) — matrix operand order repaired and checked by reading, not compiled, not run on hardware
//
#include <hip/hip_runtime.h>

#define HSZ 64
#define ISZ 3
#define BSZ 512
#define TSZ 1024

typedef _Float16 v8h  __attribute__((ext_vector_type(8)));
typedef _Float16 v16h __attribute__((ext_vector_type(16)));
typedef float    v8f  __attribute__((ext_vector_type(8)));

// Concatenate two 8-half LDS loads into one 16-half WMMA operand (VGPR order).
__device__ __forceinline__ v16h cat16(v8h lo, v8h hi) {
    return __builtin_shufflevector(lo, hi, 0,1,2,3,4,5,6,7,8,9,10,11,12,13,14,15);
}

// A-matrix fragment (16x32 f16) from a row-major LDS buffer.
// ISA 7.12.2: lanes 0-15 hold row=lane, K {0..7, 16..23}; lanes 16-31 hold
// row=lane-16, K {8..15, 24..31}.  kbase/rowHalves are in half-elements.
__device__ __forceinline__ v16h load_a(const _Float16* buf, int rowHalves,
                                       int kbase, int lc, int lh) {
    const _Float16* p = buf + lc * rowHalves + kbase + lh * 8;
    v8h lo = *(const v8h*)p;          // K = kbase + lh*8 .. +7
    v8h hi = *(const v8h*)(p + 16);   // K = kbase + 16 + lh*8 .. +7
    return cat16(lo, hi);
}

// B-matrix fragment (32x16 f16) from row-major weights [N][Ktot] in LDS.
// lanes 0-15: N=lane, K = kc*32 + {0..7, 16..23} ; lanes 16-31: N=lane-16, K = kc*32 + {8..15, 24..31}
__device__ __forceinline__ v16h load_b(const _Float16* w, int ktot, int nt,
                                       int kc, int lc, int lh) {
    const _Float16* p = w + (nt * 16 + lc) * ktot + kc * 32 + lh * 8;
    v8h lo = *(const v8h*)p;
    v8h hi = *(const v8h*)(p + 16);
    return cat16(lo, hi);
}

__device__ __forceinline__ v8f wmma_f16(v16h a, v16h b, v8f c) {
    return __builtin_amdgcn_wmma_f32_16x16x32_f16(
        /*neg_a=*/false, a, /*neg_b=*/false, b,
        /*c_mod=*/(short)0, c, /*reuse_a=*/false, /*reuse_b=*/false);
}

__device__ __forceinline__ float sigm(float x) {
    return 1.0f / (1.0f + __expf(-x));
}
__device__ __forceinline__ float tanh_fast(float x) {
    // 1 - 2/(e^{2x}+1): saturates cleanly to +/-1 without NaN at |x| large
    float e = __expf(2.0f * x);
    return 1.0f - 2.0f / (e + 1.0f);
}

__global__ __launch_bounds__(128)
void lstm2_wmma_kernel(const float* __restrict__ x,
                       const float* __restrict__ W_ih0, const float* __restrict__ W_hh0,
                       const float* __restrict__ b_ih0, const float* __restrict__ b_hh0,
                       const float* __restrict__ W_ih1, const float* __restrict__ W_hh1,
                       const float* __restrict__ b_ih1, const float* __restrict__ b_hh1,
                       const float* __restrict__ W_fc,  const float* __restrict__ b_fc,
                       float* __restrict__ out) {
    // Layer-0 combined weights, K layout: [0..2]=W_ih0, [3..31]=0, [32..95]=W_hh0
    __shared__ __align__(16) _Float16 sW0[256 * 96];    // 48 KB
    // Layer-1 combined weights, K layout: [0..63]=W_ih1 (input=h1), [64..127]=W_hh1
    __shared__ __align__(16) _Float16 sW1[256 * 128];   // 64 KB
    __shared__ __align__(16) _Float16 sH1[16 * 64];     // h1 (A operand, layer0 recurrent)
    __shared__ __align__(16) _Float16 sA1[16 * 128];    // [h1 | h2] (A operand, layer1)
    __shared__ float sB0[256];
    __shared__ float sB1[256];
    __shared__ float sWfc[64];

    const int tid  = threadIdx.x;
    const int wave = tid >> 5;
    const int lane = tid & 31;
    const int lh   = lane >> 4;   // lane half (0/1)
    const int lc   = lane & 15;   // column / row-in-half index
    const int row0 = blockIdx.x * 16;   // batch-tile base row

    // ---- one-time staging: convert weights to f16 in LDS, fold biases ----
    for (int idx = tid; idx < 256 * 96; idx += 128) {
        int n = idx / 96, k = idx - n * 96;
        float v = 0.0f;
        if (k < ISZ)       v = W_ih0[n * ISZ + k];
        else if (k >= 32)  v = W_hh0[n * HSZ + (k - 32)];
        sW0[idx] = (_Float16)v;
    }
    for (int idx = tid; idx < 256 * 128; idx += 128) {
        int n = idx >> 7, k = idx & 127;
        float v = (k < HSZ) ? W_ih1[n * HSZ + k] : W_hh1[n * HSZ + (k - HSZ)];
        sW1[idx] = (_Float16)v;
    }
    for (int n = tid; n < 256; n += 128) {
        sB0[n] = b_ih0[n] + b_hh0[n];
        sB1[n] = b_ih1[n] + b_hh1[n];
    }
    if (tid < 64) sWfc[tid] = W_fc[tid];
    for (int idx = tid; idx < 16 * 64;  idx += 128) sH1[idx] = (_Float16)0.0f;
    for (int idx = tid; idx < 16 * 128; idx += 128) sA1[idx] = (_Float16)0.0f;
    __syncthreads();

    // per-lane persistent cell state: rows r+8*lh, hidden column 16*wave+lc
    float c1r[8], c2r[8];
    #pragma unroll
    for (int r = 0; r < 8; ++r) { c1r[r] = 0.0f; c2r[r] = 0.0f; }

    const int   hidx = wave * 16 + lc;                     // hidden unit this lane owns
    const float* xrow = x + (size_t)(row0 + lc) * TSZ * ISZ;

    #pragma unroll 1
    for (int t = 0; t < TSZ; ++t) {
        // ---------------- layer 0: z0 = [x|0|h1] @ W0^T + b0 ----------------
        v8f acc[4];
        #pragma unroll
        for (int j = 0; j < 4; ++j) {
            float bv = sB0[(wave + 4 * j) * 16 + lc];
            #pragma unroll
            for (int r = 0; r < 8; ++r) acc[j][r] = bv;
        }

        // A chunk 0: x_t padded to K=32 (only lanes 0-15 carry K=0..2)
        v16h a0;
        #pragma unroll
        for (int e = 0; e < 16; ++e) a0[e] = (_Float16)0.0f;
        if (lh == 0) {
            const float* xp = xrow + t * ISZ;
            a0[0] = (_Float16)xp[0];
            a0[1] = (_Float16)xp[1];
            a0[2] = (_Float16)xp[2];
            __builtin_prefetch(xp + 8 * ISZ, 0, 0);
        }
        v16h a1 = load_a(sH1, 64, 0,  lc, lh);   // h1[0..31]
        v16h a2 = load_a(sH1, 64, 32, lc, lh);   // h1[32..63]

        #pragma unroll
        for (int j = 0; j < 4; ++j) {
            int nt = wave + 4 * j;  // tiles j=0..3 -> gates i,f,g,o for hidden slice
            acc[j] = wmma_f16(a0, load_b(sW0, 96, nt, 0, lc, lh), acc[j]);
            acc[j] = wmma_f16(a1, load_b(sW0, 96, nt, 1, lc, lh), acc[j]);
            acc[j] = wmma_f16(a2, load_b(sW0, 96, nt, 2, lc, lh), acc[j]);
        }

        // element-wise cell update (registers only)
        float h1v[8];
        #pragma unroll
        for (int r = 0; r < 8; ++r) {
            float iv = sigm(acc[0][r]);
            float fv = sigm(acc[1][r]);
            float gv = tanh_fast(acc[2][r]);
            float ov = sigm(acc[3][r]);
            float cn = fv * c1r[r] + iv * gv;
            c1r[r] = cn;
            h1v[r] = ov * tanh_fast(cn);
        }
        __syncthreads();   // B1: all layer-0 reads of sH1 done -> safe to overwrite

        #pragma unroll
        for (int r = 0; r < 8; ++r) {
            int row = r + 8 * lh;
            _Float16 hv = (_Float16)h1v[r];
            sH1[row * 64  + hidx] = hv;   // recurrent input for next step
            sA1[row * 128 + hidx] = hv;   // layer-1 input (K 0..63)
        }
        __syncthreads();   // B2: h1 visible to everyone

        // ---------------- layer 1: z1 = [h1|h2] @ W1^T + b1 ----------------
        v8f acc2[4];
        #pragma unroll
        for (int j = 0; j < 4; ++j) {
            float bv = sB1[(wave + 4 * j) * 16 + lc];
            #pragma unroll
            for (int r = 0; r < 8; ++r) acc2[j][r] = bv;
        }
        v16h aa0 = load_a(sA1, 128, 0,  lc, lh);
        v16h aa1 = load_a(sA1, 128, 32, lc, lh);
        v16h aa2 = load_a(sA1, 128, 64, lc, lh);
        v16h aa3 = load_a(sA1, 128, 96, lc, lh);
        #pragma unroll
        for (int j = 0; j < 4; ++j) {
            int nt = wave + 4 * j;
            acc2[j] = wmma_f16(aa0, load_b(sW1, 128, nt, 0, lc, lh), acc2[j]);
            acc2[j] = wmma_f16(aa1, load_b(sW1, 128, nt, 1, lc, lh), acc2[j]);
            acc2[j] = wmma_f16(aa2, load_b(sW1, 128, nt, 2, lc, lh), acc2[j]);
            acc2[j] = wmma_f16(aa3, load_b(sW1, 128, nt, 3, lc, lh), acc2[j]);
        }

        float h2v[8];
        #pragma unroll
        for (int r = 0; r < 8; ++r) {
            float iv = sigm(acc2[0][r]);
            float fv = sigm(acc2[1][r]);
            float gv = tanh_fast(acc2[2][r]);
            float ov = sigm(acc2[3][r]);
            float cn = fv * c2r[r] + iv * gv;
            c2r[r] = cn;
            h2v[r] = ov * tanh_fast(cn);
        }
        __syncthreads();   // B3: layer-1 reads of sA1 done -> safe to overwrite h2

        #pragma unroll
        for (int r = 0; r < 8; ++r) {
            int row = r + 8 * lh;
            sA1[row * 128 + 64 + hidx] = (_Float16)h2v[r];   // layer-1 input (K 64..127)
        }
    }
    __syncthreads();

    // ---------------- head: out = sigmoid(h2 . W_fc + b_fc) ----------------
    if (tid < 16) {
        float z = b_fc[0];
        #pragma unroll 8
        for (int k = 0; k < HSZ; ++k)
            z += (float)sA1[tid * 128 + 64 + k] * sWfc[k];
        out[row0 + tid] = 1.0f / (1.0f + __expf(-z));
    }
}

extern "C" void kernel_launch(void* const* d_in, const int* in_sizes, int n_in,
                              void* d_out, int out_size, void* d_ws, size_t ws_size,
                              hipStream_t stream) {
    (void)in_sizes; (void)n_in; (void)d_ws; (void)ws_size; (void)out_size;
    const float* xx    = (const float*)d_in[0];
    const float* Wih0  = (const float*)d_in[1];
    const float* Whh0  = (const float*)d_in[2];
    const float* bih0  = (const float*)d_in[3];
    const float* bhh0  = (const float*)d_in[4];
    const float* Wih1  = (const float*)d_in[5];
    const float* Whh1  = (const float*)d_in[6];
    const float* bih1  = (const float*)d_in[7];
    const float* bhh1  = (const float*)d_in[8];
    const float* Wfc   = (const float*)d_in[9];
    const float* bfc   = (const float*)d_in[10];
    float* out = (float*)d_out;

    lstm2_wmma_kernel<<<dim3(BSZ / 16), dim3(128), 0, stream>>>(
        xx, Wih0, Whh0, bih0, bhh0, Wih1, Whh1, bih1, bhh1, Wfc, bfc, out);
}
